// VQLayer_25984552141247
// MI455X (gfx1250) — matrix operand order repaired and checked by reading, not compiled, not run on hardware
//
#include <hip/hip_runtime.h>
#include <stdint.h>

typedef __attribute__((ext_vector_type(16))) __bf16 v16bf;
typedef __attribute__((ext_vector_type(8)))  float  v8f;

#define CODEBOOK_K 8192
#define DIM        256
#define NROWS      32768            // 512 * 64
#define BETA       0.25f
#define NWAVES     (NROWS / 16)     // 2048 waves, 16 rows each
#define TILE_C     32               // codes staged per LDS tile
#define NTILES     (CODEBOOK_K / TILE_C)   // 256 iterations

// workspace layout (bytes)
#define WS_EHI    0u                        // 8192*256*2 = 4 MB  bf16 hi
#define WS_ELO    4194304u                  // 4 MB               bf16 lo
#define WS_ENORM  8388608u                  // 8192*4             ||e||^2
#define WS_COUNT  8421376u                  // 8192*4             usage counts
#define WS_LOSS   8454144u                  // 2048*4             per-wave loss

static __device__ __forceinline__ __bf16 to_bf16(float f) { return (__bf16)f; }

// One per-lane 16-byte async global->LDS transfer (ASYNCcnt-tracked).
// VDST vgpr = LDS byte address (low 32 bits of generic shared pointer,
// per ISA 10.2 aperture truncation), VADDR = 64-bit global address.
#define ASYNC_CP_B128(ldsoff, gptr)                                        \
    asm volatile("global_load_async_to_lds_b128 %0, %1, off"               \
                 :: "v"(ldsoff), "v"(gptr) : "memory")

#define WAIT_ASYNCCNT_8() asm volatile("s_wait_asynccnt 0x8" ::: "memory")
#define WAIT_ASYNCCNT_0() asm volatile("s_wait_asynccnt 0x0" ::: "memory")

// ---------------------------------------------------------------------------
// Kernel 1: codebook -> bf16 hi/lo split, ||e||^2, zero counters
// ---------------------------------------------------------------------------
__global__ void __launch_bounds__(256)
vq_prep(const float* __restrict__ embed, __bf16* __restrict__ e_hi,
        __bf16* __restrict__ e_lo, float* __restrict__ enorm,
        int* __restrict__ counts, float* __restrict__ lossbuf) {
    const int lane = threadIdx.x & 31;
    const int wave = threadIdx.x >> 5;
    const int code = blockIdx.x * 8 + wave;

    const int gid = blockIdx.x * 256 + threadIdx.x;
    if (gid < CODEBOOK_K) counts[gid] = 0;
    if (gid < NWAVES)     lossbuf[gid] = 0.0f;

    const float* row = embed + (size_t)code * DIM;
    float nrm = 0.0f;
#pragma unroll
    for (int j = 0; j < 8; ++j) {
        const int d = j * 32 + lane;              // coalesced
        const float v = row[d];
        const __bf16 h = to_bf16(v);
        const __bf16 l = to_bf16(v - (float)h);
        e_hi[(size_t)code * DIM + d] = h;
        e_lo[(size_t)code * DIM + d] = l;
        nrm += v * v;
    }
#pragma unroll
    for (int m = 16; m >= 1; m >>= 1) nrm += __shfl_xor(nrm, m, 32);
    if (lane == 0) enorm[code] = nrm;
}

// ---------------------------------------------------------------------------
// Kernel 2: fused GEMM-distance + argmin + gather + loss partials.
// Block = 8 waves = 128 rows. 32-code B tiles (hi+lo = 32 KB, contiguous in
// memory) staged per block into LDS by async copies, double buffered; tile
// t+1 streams while tile t feeds 48 bf16 WMMAs/wave. B-fragment LDS loads
// are software-pipelined one k-chunk ahead of their WMMAs.
// ---------------------------------------------------------------------------
__global__ void __launch_bounds__(256)
vq_main(const float* __restrict__ x, const float* __restrict__ embed,
        const __bf16* __restrict__ e_hi, const __bf16* __restrict__ e_lo,
        const float* __restrict__ enorm, int* __restrict__ counts,
        float* __restrict__ lossbuf, float* __restrict__ out) {
    // LDS: [buffer][hi/lo][32 codes * 256 k] bf16 = 64 KB total
    __shared__ __bf16 sh[2][2][TILE_C * DIM];

    const int tid    = threadIdx.x;
    const int lane   = tid & 31;
    const int wib    = tid >> 5;                    // wave in block, 0..7
    const int waveId = blockIdx.x * 8 + wib;        // 0..2047
    const int mbase  = waveId * 16;
    const int hf     = lane >> 4;                   // 16-lane half
    const int lm     = lane & 15;

    // ---- Load A tile (16 rows x 256 k) in WMMA 16-bit A layout, hi/lo split.
    // 16-bit A 16x32: k(v,e,half) = half*16 + v*2 + e (contiguous, matches B)
    v16bf a_hi[8], a_lo[8];
    {
        const float* xrow = x + (size_t)(mbase + lm) * DIM;
#pragma unroll
        for (int c = 0; c < 8; ++c) {
#pragma unroll
            for (int v = 0; v < 8; ++v) {
                const int k = c * 32 + hf * 16 + v * 2;
                const float f0 = xrow[k], f1 = xrow[k + 1];
                const __bf16 h0 = to_bf16(f0), h1 = to_bf16(f1);
                a_hi[c][2 * v]     = h0;
                a_hi[c][2 * v + 1] = h1;
                a_lo[c][2 * v]     = to_bf16(f0 - (float)h0);
                a_lo[c][2 * v + 1] = to_bf16(f1 - (float)h1);
            }
        }
    }

    // Per-thread async-copy slices: tile occupies 16 KB each in e_hi / e_lo.
    // 256 threads x 16 B = 4 KB per instruction round -> 8 instrs/thread/tile.
    const uint32_t lds_hi0 = (uint32_t)(uintptr_t)&sh[0][0][0];
    const uint32_t lds_lo0 = (uint32_t)(uintptr_t)&sh[0][1][0];
    const uint32_t lds_hi1 = (uint32_t)(uintptr_t)&sh[1][0][0];
    const uint32_t lds_lo1 = (uint32_t)(uintptr_t)&sh[1][1][0];
    const uint32_t so = (uint32_t)tid * 16u;        // byte slice in tile

#define PREFETCH_TILE(T, LHI, LLO)                                          \
    do {                                                                    \
        const char* gh = (const char*)e_hi + (size_t)(T) * 16384 + so;      \
        const char* gl = (const char*)e_lo + (size_t)(T) * 16384 + so;      \
        ASYNC_CP_B128((LHI) + so,          gh);                             \
        ASYNC_CP_B128((LHI) + so + 4096u,  gh + 4096);                      \
        ASYNC_CP_B128((LHI) + so + 8192u,  gh + 8192);                      \
        ASYNC_CP_B128((LHI) + so + 12288u, gh + 12288);                     \
        ASYNC_CP_B128((LLO) + so,          gl);                             \
        ASYNC_CP_B128((LLO) + so + 4096u,  gl + 4096);                      \
        ASYNC_CP_B128((LLO) + so + 8192u,  gl + 8192);                      \
        ASYNC_CP_B128((LLO) + so + 12288u, gl + 12288);                     \
    } while (0)

    PREFETCH_TILE(0, lds_hi0, lds_lo0);

    float best[8];
    int   bidx[8];
#pragma unroll
    for (int r = 0; r < 8; ++r) { best[r] = 3.4e38f; bidx[r] = 0; }

    for (int t = 0; t < NTILES; ++t) {
        const int cur = t & 1;
        if (t + 1 < NTILES) {
            if (cur) PREFETCH_TILE(t + 1, lds_hi0, lds_lo0);
            else     PREFETCH_TILE(t + 1, lds_hi1, lds_lo1);
            WAIT_ASYNCCNT_8();              // drain tile t, keep t+1 in flight
        } else {
            WAIT_ASYNCCNT_0();
        }
        __syncthreads();                     // tile t visible to all 8 waves

#pragma unroll
        for (int sub = 0; sub < 2; ++sub) {
            const int col = t * TILE_C + sub * 16 + lm;
            const float en = enorm[col];     // overlaps with WMMAs below
            // B 32x16 in this code's contiguous k order (same as A): lane = column, lanes 0-15 hold k 0-15,
            // lanes 16-31 hold k 16-31 -> contiguous 16 bf16 along D in LDS.
            const __bf16* bh = &sh[cur][0][(sub * 16 + lm) * DIM + hf * 16];
            const __bf16* bl = &sh[cur][1][(sub * 16 + lm) * DIM + hf * 16];
            v16bf bhi = *(const v16bf*)(bh);
            v16bf blo = *(const v16bf*)(bl);
            v8f acc = {};
#pragma unroll
            for (int c = 0; c < 8; ++c) {
                v16bf nhi, nlo;              // pipeline: fetch chunk c+1
                if (c < 7) {
                    nhi = *(const v16bf*)(bh + (c + 1) * 32);
                    nlo = *(const v16bf*)(bl + (c + 1) * 32);
                }
                acc = __builtin_amdgcn_wmma_f32_16x16x32_bf16(false, a_hi[c], false, bhi, (short)0, acc, false, false);
                acc = __builtin_amdgcn_wmma_f32_16x16x32_bf16(false, a_lo[c], false, bhi, (short)0, acc, false, false);
                acc = __builtin_amdgcn_wmma_f32_16x16x32_bf16(false, a_hi[c], false, blo, (short)0, acc, false, false);
                if (c < 7) { bhi = nhi; blo = nlo; }
            }
#pragma unroll
            for (int r = 0; r < 8; ++r) {
                const float d = en - 2.0f * acc[r];  // ||z||^2 const -> dropped
                if (d < best[r]) { best[r] = d; bidx[r] = col; }
            }
        }
        __syncthreads();                     // safe to overwrite buf[cur] next
    }
#undef PREFETCH_TILE

    // ---- Cross-lane argmin inside each 16-lane group (rows 0-7 | 8-15).
#pragma unroll
    for (int m = 8; m >= 1; m >>= 1) {
#pragma unroll
        for (int r = 0; r < 8; ++r) {
            const float od = __shfl_xor(best[r], m, 32);
            const int   oi = __shfl_xor(bidx[r], m, 32);
            if (od < best[r] || (od == best[r] && oi < bidx[r])) {
                best[r] = od; bidx[r] = oi;
            }
        }
    }

    // lanes 0 / 16 now hold rows mbase+0..7 / mbase+8..15
    if (lm == 0) {
#pragma unroll
        for (int r = 0; r < 8; ++r) {
            const int row = mbase + hf * 8 + r;
            const int idx = bidx[r];
            out[(size_t)NROWS * DIM + 2 + row] = (float)idx;  // embed_ind
            atomicAdd(&counts[idx], 1);                        // int: deterministic
        }
    }

    // ---- Gather x_q, emit straight-through output, accumulate loss partial.
    float lacc = 0.0f;
#pragma unroll
    for (int r = 0; r < 8; ++r) {
        const int i0 = __shfl(bidx[r], 0, 32);
        const int i1 = __shfl(bidx[r], 16, 32);
        const int rows[2] = { mbase + r, mbase + 8 + r };
        const int idxs[2] = { i0, i1 };
#pragma unroll
        for (int s = 0; s < 2; ++s) {
            const float* erow = embed + (size_t)idxs[s] * DIM;
            const float* xrow = x + (size_t)rows[s] * DIM;
            float* orow = out + (size_t)rows[s] * DIM;
#pragma unroll
            for (int j = 0; j < 8; ++j) {
                const int d = j * 32 + lane;
                const float xq = erow[d];
                const float xv = xrow[d];
                const float diff = xq - xv;
                orow[d] = xv + diff;              // x + sg(x_q - x)
                lacc += diff * diff;
            }
        }
    }
#pragma unroll
    for (int m = 16; m >= 1; m >>= 1) lacc += __shfl_xor(lacc, m, 32);
    if (lane == 0) lossbuf[waveId] = lacc;        // no float atomics
}

// ---------------------------------------------------------------------------
// Kernel 3: fixed-order scalar finalization (loss, unused-code count).
// ---------------------------------------------------------------------------
__global__ void __launch_bounds__(256)
vq_final(const int* __restrict__ counts, const float* __restrict__ lossbuf,
         float* __restrict__ out) {
    __shared__ float sl[256];
    __shared__ int   su[256];
    const int t = threadIdx.x;
    float ls = 0.0f;
    for (int i = t; i < NWAVES; i += 256) ls += lossbuf[i];
    int uz = 0;
    for (int i = t; i < CODEBOOK_K; i += 256) uz += (counts[i] == 0);
    sl[t] = ls; su[t] = uz;
    __syncthreads();
    for (int s = 128; s > 0; s >>= 1) {
        if (t < s) { sl[t] += sl[t + s]; su[t] += su[t + s]; }
        __syncthreads();
    }
    if (t == 0) {
        out[(size_t)NROWS * DIM]     = sl[0] * (1.0f + BETA) / (float)((size_t)NROWS * DIM);
        out[(size_t)NROWS * DIM + 1] = (float)su[0];
    }
}

// ---------------------------------------------------------------------------
extern "C" void kernel_launch(void* const* d_in, const int* in_sizes, int n_in,
                              void* d_out, int out_size, void* d_ws, size_t ws_size,
                              hipStream_t stream) {
    const float* x     = (const float*)d_in[0];   // [512,64,256]
    const float* embed = (const float*)d_in[1];   // [8192,256]
    char* ws = (char*)d_ws;
    __bf16* e_hi   = (__bf16*)(ws + WS_EHI);
    __bf16* e_lo   = (__bf16*)(ws + WS_ELO);
    float*  enorm  = (float*)(ws + WS_ENORM);
    int*    counts = (int*)(ws + WS_COUNT);
    float*  lossb  = (float*)(ws + WS_LOSS);
    float*  out    = (float*)d_out;

    vq_prep<<<CODEBOOK_K / 8, 256, 0, stream>>>(embed, e_hi, e_lo, enorm, counts, lossb);
    vq_main<<<NWAVES / 8, 256, 0, stream>>>(x, embed, e_hi, e_lo, enorm, counts, lossb, out);
    vq_final<<<1, 256, 0, stream>>>(counts, lossb, out);
}
